// GraphNet_73194832658843
// MI455X (gfx1250) — matrix operand order repaired and checked by reading, not compiled, not run on hardware
//
#include <hip/hip_runtime.h>
#include <math.h>

// ---------------- types / helpers ----------------
typedef __attribute__((ext_vector_type(16))) __bf16 v16bf;
typedef __attribute__((ext_vector_type(8)))  float  v8f;

#define DEVINL __device__ __forceinline__

DEVINL unsigned short f2bf(float f) {           // fp32 -> bf16, round-nearest-even
  unsigned int u = __float_as_uint(f);
  unsigned int r = u + 0x7fffu + ((u >> 16) & 1u);
  return (unsigned short)(r >> 16);
}

// ---------------- problem constants ----------------
#define NB    32
#define N0    2048
#define FEATD 1036
#define NHID  128
#define EDG   1048576          /* 32*2048*16 */
#define KK1   410
#define KK2   82
#define KK3   17
#define NN1   (NB*N0)          /* 65536 */
#define NKK1  (NB*KK1)         /* 13120 */
#define NKK2  (NB*KK2)         /* 2624  */
#define NKK3  (NB*KK3)         /* 544   */
#define KP1   1056             /* 1036 padded to 32 */
#define KT1   33

// =====================================================================
// GEMM: C = A(f32,row-major,[M,KDIM]) x B(packed bf16); bf16 WMMA, f32 acc.
// One wave per block; wave computes a 16 x (NTILES*16) strip. All loop
// bounds are template constants so accumulators stay in fixed VGPRs
// (no movrel indexing) and the k-tail is a constexpr epilogue.
// DUAL: second weight matrix shares the A registers (halves A traffic).
// USESCL: per-k scale (feature normalization folded into the A load).
// =====================================================================
template<int NTILES, bool DUAL>
DEVINL void wmma_step(const v16bf a, int tilebase, int lane,
                      const unsigned short* __restrict__ B1,
                      const unsigned short* __restrict__ B2,
                      v8f* acc1, v8f* acc2)
{
  #pragma unroll
  for (int t = 0; t < NTILES; ++t) {
    const size_t tb = ((size_t)(tilebase + t) * 32 + lane) * 16;
    v16bf b = *(const v16bf*)(B1 + tb);
    acc1[t] = __builtin_amdgcn_wmma_f32_16x16x32_bf16(
                  false, a, false, b, (short)0, acc1[t], false, false);
    if constexpr (DUAL) {
      v16bf b2 = *(const v16bf*)(B2 + tb);
      acc2[t] = __builtin_amdgcn_wmma_f32_16x16x32_bf16(
                  false, a, false, b2, (short)0, acc2[t], false, false);
    }
  }
}

template<int KDIM, int NTILES, bool DUAL, bool USESCL>
__global__ __launch_bounds__(32)
void k_gemm(const float* __restrict__ A, const float* __restrict__ scl,
            const unsigned short* __restrict__ B1,
            const unsigned short* __restrict__ B2,
            float* __restrict__ C1, float* __restrict__ C2,
            int N, int NTtot)
{
  constexpr int KFULL = KDIM / 32;             // full k-tiles
  const int lane  = threadIdx.x & 31;
  const int mt    = blockIdx.x;
  const int nt0   = blockIdx.y * NTILES;

  const int mrow  = mt * 16 + (lane & 15);
  const int khalf = lane >> 4;                 // 0 or 1
  const float* Arow = A + (size_t)mrow * KDIM;

  v8f acc1[NTILES], acc2[DUAL ? NTILES : 1];
  #pragma unroll
  for (int t = 0; t < NTILES; ++t)
    #pragma unroll
    for (int i = 0; i < 8; ++i) { acc1[t][i] = 0.f; if constexpr (DUAL) acc2[t][i] = 0.f; }

  for (int kt = 0; kt < KFULL; ++kt) {
    const int kb = kt * 32 + khalf * 8;
    const float4 p0 = *(const float4*)(Arow + kb);
    const float4 p1 = *(const float4*)(Arow + kb + 4);
    const float4 p2 = *(const float4*)(Arow + kb + 16);
    const float4 p3 = *(const float4*)(Arow + kb + 20);
    float af[16] = { p0.x,p0.y,p0.z,p0.w, p1.x,p1.y,p1.z,p1.w,
                     p2.x,p2.y,p2.z,p2.w, p3.x,p3.y,p3.z,p3.w };
    if constexpr (USESCL) {
      #pragma unroll
      for (int j = 0; j < 16; ++j) { int k = kb + j + ((j >= 8) ? 8 : 0); af[j] *= scl[k]; }
    }
    union { v16bf v; unsigned short s[16]; } ua;
    #pragma unroll
    for (int j = 0; j < 16; ++j) ua.s[j] = f2bf(af[j]);
    wmma_step<NTILES, DUAL>(ua.v, kt * NTtot + nt0, lane, B1, B2, acc1, acc2);
  }

  if constexpr (KDIM % 32 != 0) {              // k-tail (layer 1: K=1036)
    const int kb = KFULL * 32 + khalf * 8;
    float af[16];
    #pragma unroll
    for (int j = 0; j < 8; ++j) { int k = kb + j;      af[j]   = (k < KDIM) ? Arow[k] : 0.f; }
    #pragma unroll
    for (int j = 0; j < 8; ++j) { int k = kb + 16 + j; af[8+j] = (k < KDIM) ? Arow[k] : 0.f; }
    if constexpr (USESCL) {
      #pragma unroll
      for (int j = 0; j < 16; ++j) { int k = kb + j + ((j >= 8) ? 8 : 0); af[j] *= scl[k]; }
    }
    union { v16bf v; unsigned short s[16]; } ua;
    #pragma unroll
    for (int j = 0; j < 16; ++j) ua.s[j] = f2bf(af[j]);
    wmma_step<NTILES, DUAL>(ua.v, KFULL * NTtot + nt0, lane, B1, B2, acc1, acc2);
  }

  // C/D layout: VGPR r -> M = khalf*8 + r ; N = (lane&15)
  const int mbase = mt * 16 + khalf * 8;
  const int ncol  = lane & 15;
  #pragma unroll
  for (int t = 0; t < NTILES; ++t) {
    const int n = (nt0 + t) * 16 + ncol;
    #pragma unroll
    for (int r = 0; r < 8; ++r) {
      C1[(size_t)(mbase + r) * N + n] = acc1[t][r];
      if constexpr (DUAL) C2[(size_t)(mbase + r) * N + n] = acc2[t][r];
    }
  }
}

// Pack W[K,N] f32 row-major into WMMA B layout (bf16), zero-padded K.
__global__ void k_pack(const float* __restrict__ W, int K, int N, int NT,
                       unsigned short* __restrict__ P, int total)
{
  int t = blockIdx.x * blockDim.x + threadIdx.x;
  if (t >= total) return;
  int j    = t & 15;
  int lane = (t >> 4) & 31;
  int tile = t >> 9;
  int nt = tile % NT, kt = tile / NT;
  int k = kt * 32 + ((lane >> 4) << 3) + j + ((j >> 3) << 3);    // lane<16: K 0..7,16..23 ; lane>=16: K 8..15,24..31
  int n = nt * 16 + (lane & 15);
  float v = (k < K) ? W[(size_t)k * N + n] : 0.f;
  P[t] = f2bf(v);
}

// ---------------- elementwise / graph kernels ----------------
__global__ void k_fillf(float* p, float v, int n)
{ int i = blockIdx.x * blockDim.x + threadIdx.x; if (i < n) p[i] = v; }

__global__ void k_filli(int* p, int v, int n)
{ int i = blockIdx.x * blockDim.x + threadIdx.x; if (i < n) p[i] = v; }

__global__ void k_colmax(const float* __restrict__ x, int rows, int stride, float* __restrict__ CM)
{
  __shared__ float red[256];
  int c = blockIdx.x;
  float m = -3.402823466e38f;
  for (int r = threadIdx.x; r < rows; r += blockDim.x)
    m = fmaxf(m, x[(size_t)r * stride + c]);
  red[threadIdx.x] = m; __syncthreads();
  for (int s = 128; s > 0; s >>= 1) {
    if (threadIdx.x < s) red[threadIdx.x] = fmaxf(red[threadIdx.x], red[threadIdx.x + s]);
    __syncthreads();
  }
  if (threadIdx.x == 0) CM[c] = red[0];
}

__global__ void k_scl(const float* __restrict__ CM, float* __restrict__ S, int kp)
{ int i = blockIdx.x * blockDim.x + threadIdx.x; if (i < kp) S[i] = (i < 12) ? 1.f / CM[i] : 1.f; }

// mean aggregation: 32 lanes per edge, 4 dims/lane, f32 atomics into L2
__global__ void k_edge_agg(const float* __restrict__ H, const int* __restrict__ src,
                           const int* __restrict__ dst, const float* __restrict__ em,
                           float* __restrict__ S, float* __restrict__ CNT, int E)
{
  int t = blockIdx.x * blockDim.x + threadIdx.x;
  int e = t >> 5;
  if (e >= E) return;
  float w = em[e];
  if (w == 0.f) return;
  int c = t & 31;
  int s = src[e], d = dst[e];
  const float* hs = H + (size_t)s * NHID;
  float* sd = S + (size_t)d * NHID;
  #pragma unroll
  for (int j = 0; j < 4; ++j) atomicAdd(&sd[c * 4 + j], hs[c * 4 + j] * w);
  if (c == 0) atomicAdd(&CNT[d], w);
}

// x = relu(S/max(cnt,1) + b + R)  (written back into S)
__global__ void k_combine(float* __restrict__ S, const float* __restrict__ CNT,
                          const float* __restrict__ b, const float* __restrict__ R, int total)
{
  int i = blockIdx.x * blockDim.x + threadIdx.x;
  if (i >= total) return;
  float c = CNT[i >> 7];
  float v = S[i] / fmaxf(c, 1.f) + b[i & 127] + R[i];
  S[i] = fmaxf(v, 0.f);
}

__global__ void k_dot128(const float* __restrict__ X, const float* __restrict__ w,
                         float* __restrict__ out, int n)
{
  int i = blockIdx.x * blockDim.x + threadIdx.x;
  if (i >= n) return;
  const float* r = X + (size_t)i * NHID;
  float a = 0.f;
  #pragma unroll 8
  for (int j = 0; j < NHID; ++j) a += r[j] * w[j];
  out[i] = a;
}

__global__ void k_edge_deg(const int* __restrict__ dst, const float* __restrict__ em,
                           float* __restrict__ DEGv, int E)
{
  int e = blockIdx.x * blockDim.x + threadIdx.x;
  if (e >= E) return;
  float w = em[e];
  if (w != 0.f) atomicAdd(&DEGv[dst[e]], w);
}

__global__ void k_edge_sc(const float* __restrict__ HS, const int* __restrict__ src,
                          const int* __restrict__ dst, const float* __restrict__ em,
                          const float* __restrict__ DEGv, float* __restrict__ AGG, int E)
{
  int e = blockIdx.x * blockDim.x + threadIdx.x;
  if (e >= E) return;
  float w = em[e];
  if (w == 0.f) return;
  int s = src[e], d = dst[e];
  float norm = rsqrtf(DEGv[s]) * rsqrtf(DEGv[d]) * w;
  atomicAdd(&AGG[d], HS[s] * norm);
}

__global__ void k_score_fin(const float* __restrict__ AGG, const float* __restrict__ HS,
                            const float* __restrict__ DEGv, const float* __restrict__ bp,
                            float* __restrict__ SC, int n)
{
  int i = blockIdx.x * blockDim.x + threadIdx.x;
  if (i >= n) return;
  SC[i] = AGG[i] + HS[i] / DEGv[i] + bp[0];
}

// exact jax.lax.top_k semantics via rank (desc value, tie -> lower index)
__global__ void k_topk(const float* __restrict__ SC, int n_per, int k,
                       int* __restrict__ PERM, int n_total)
{
  int i = blockIdx.x * blockDim.x + threadIdx.x;
  if (i >= n_total) return;
  int g  = i / n_per;
  int li = i - g * n_per;
  const float* s = SC + (size_t)g * n_per;
  float si = s[li];
  int rank = 0;
  for (int j = 0; j < n_per; ++j) {
    float sj = s[j];
    rank += (sj > si) || (sj == si && j < li);
  }
  if (rank < k) PERM[g * k + rank] = i;
}

__global__ void k_scatter(const int* __restrict__ PERM, int* __restrict__ INV, int nk)
{ int p = blockIdx.x * blockDim.x + threadIdx.x; if (p < nk) INV[PERM[p]] = p; }

__global__ void k_pool(const float* __restrict__ X, const float* __restrict__ SC,
                       const int* __restrict__ PERM, float* __restrict__ XP, int total)
{
  int t = blockIdx.x * blockDim.x + threadIdx.x;
  if (t >= total) return;
  int p = t >> 7, d = t & 127;
  int node = PERM[p];
  XP[t] = X[(size_t)node * NHID + d] * tanhf(SC[node]);
}

__global__ void k_remap(const int* __restrict__ src, const int* __restrict__ dst,
                        const float* __restrict__ em, const int* __restrict__ INV,
                        int* __restrict__ ns, int* __restrict__ nd,
                        float* __restrict__ ne, int E)
{
  int e = blockIdx.x * blockDim.x + threadIdx.x;
  if (e >= E) return;
  int a = INV[src[e]], b = INV[dst[e]];
  bool ok = (a >= 0) && (b >= 0);
  ns[e] = a >= 0 ? a : 0;
  nd[e] = b >= 0 ? b : 0;
  ne[e] = ok ? em[e] : 0.f;
}

__global__ void k_readout(const float* __restrict__ XP, int k, float* __restrict__ HSUM, int total)
{
  int t = blockIdx.x * blockDim.x + threadIdx.x;
  if (t >= total) return;
  int g = t >> 7, d = t & 127;
  float mx = -3.402823466e38f, sm = 0.f;
  for (int r = 0; r < k; ++r) {
    float v = XP[((size_t)(g * k + r)) * NHID + d];
    mx = fmaxf(mx, v); sm += v;
  }
  HSUM[g * 256 + d]       += mx;
  HSUM[g * 256 + 128 + d] += sm / (float)k;
}

__global__ void k_bias_relu(float* __restrict__ X, const float* __restrict__ b, int total, int N)
{
  int t = blockIdx.x * blockDim.x + threadIdx.x;
  if (t >= total) return;
  X[t] = fmaxf(X[t] + b[t % N], 0.f);
}

__global__ void k_final(const float* __restrict__ F, const float* __restrict__ w,
                        const float* __restrict__ b, float* __restrict__ out)
{
  int g = threadIdx.x;              // 32 graphs
  float a = 0.f;
  #pragma unroll
  for (int j = 0; j < 32; ++j) a += F[g * 32 + j] * w[j];
  out[g] = a + b[0];
}

// =====================================================================
#define CDIV(a, b) (((a) + (b) - 1) / (b))
#define EL(krn, n, ...) krn<<<CDIV((n), 256), 256, 0, stream>>>(__VA_ARGS__)

extern "C" void kernel_launch(void* const* d_in, const int* in_sizes, int n_in,
                              void* d_out, int out_size, void* d_ws, size_t ws_size,
                              hipStream_t stream)
{
  const float* x    = (const float*)d_in[0];
  const int*   src0 = (const int*)  d_in[1];
  const int*   dst0 = (const int*)  d_in[2];
  const float* W1l  = (const float*)d_in[3];
  const float* b1l  = (const float*)d_in[4];
  const float* W1r  = (const float*)d_in[5];
  const float* Wp1  = (const float*)d_in[6];
  const float* bp1  = (const float*)d_in[7];
  const float* W2l  = (const float*)d_in[8];
  const float* b2l  = (const float*)d_in[9];
  const float* W2r  = (const float*)d_in[10];
  const float* Wp2  = (const float*)d_in[11];
  const float* bp2  = (const float*)d_in[12];
  const float* W3l  = (const float*)d_in[13];
  const float* b3l  = (const float*)d_in[14];
  const float* W3r  = (const float*)d_in[15];
  const float* Wp3  = (const float*)d_in[16];
  const float* bp3  = (const float*)d_in[17];
  const float* l1w  = (const float*)d_in[18];
  const float* l1b  = (const float*)d_in[19];
  const float* l2w  = (const float*)d_in[20];
  const float* l2b  = (const float*)d_in[21];
  const float* l3w  = (const float*)d_in[22];
  const float* l3b  = (const float*)d_in[23];
  float* out = (float*)d_out;

  // ---- carve workspace (all offsets 256B-aligned) ----
  char* wp = (char*)d_ws;
  auto carve = [&](size_t bytes) -> char* {
    char* p = wp; wp += (bytes + 255) & ~(size_t)255; return p;
  };
  float* H    = (float*)carve((size_t)NN1 * NHID * 4);
  float* R    = (float*)carve((size_t)NN1 * NHID * 4);
  float* S    = (float*)carve((size_t)NN1 * NHID * 4);
  float* CNT  = (float*)carve((size_t)NN1 * 4);
  float* DEGv = (float*)carve((size_t)NN1 * 4);
  float* HS   = (float*)carve((size_t)NN1 * 4);
  float* AGG  = (float*)carve((size_t)NN1 * 4);
  float* SCR  = (float*)carve((size_t)NN1 * 4);
  int*   INV  = (int*)  carve((size_t)NN1 * 4);
  int*   PERM = (int*)  carve((size_t)NKK1 * 4);
  float* EM0  = (float*)carve((size_t)EDG * 4);
  int*   SRCA = (int*)  carve((size_t)EDG * 4);
  int*   DSTA = (int*)  carve((size_t)EDG * 4);
  float* EMA  = (float*)carve((size_t)EDG * 4);
  int*   SRCB = (int*)  carve((size_t)EDG * 4);
  int*   DSTB = (int*)  carve((size_t)EDG * 4);
  float* EMB  = (float*)carve((size_t)EDG * 4);
  float* XP1  = (float*)carve((size_t)NKK1 * NHID * 4);
  float* XP2  = (float*)carve((size_t)NKK2 * NHID * 4);
  float* XP3  = (float*)carve((size_t)NKK3 * NHID * 4);
  float* HSUM = (float*)carve((size_t)NB * 256 * 4);
  float* H1   = (float*)carve((size_t)NB * NHID * 4);
  float* CM   = (float*)carve(64);
  float* SCL  = (float*)carve((size_t)KP1 * 4);
  unsigned short* PW1L = (unsigned short*)carve((size_t)KT1 * 8 * 512 * 2);
  unsigned short* PW1R = (unsigned short*)carve((size_t)KT1 * 8 * 512 * 2);
  unsigned short* PW2L = (unsigned short*)carve((size_t)4 * 8 * 512 * 2);
  unsigned short* PW2R = (unsigned short*)carve((size_t)4 * 8 * 512 * 2);
  unsigned short* PW3L = (unsigned short*)carve((size_t)4 * 8 * 512 * 2);
  unsigned short* PW3R = (unsigned short*)carve((size_t)4 * 8 * 512 * 2);
  unsigned short* PL1  = (unsigned short*)carve((size_t)8 * 8 * 512 * 2);
  unsigned short* PL2  = (unsigned short*)carve((size_t)4 * 2 * 512 * 2);

  // ---- pack all weights into WMMA B layout ----
  auto pack = [&](const float* W, int K, int N, int KT, unsigned short* P) {
    int NT = N / 16, total = KT * NT * 512;
    EL(k_pack, total, W, K, N, NT, P, total);
  };
  pack(W1l, FEATD, NHID, KT1, PW1L);  pack(W1r, FEATD, NHID, KT1, PW1R);
  pack(W2l, NHID,  NHID, 4,   PW2L);  pack(W2r, NHID,  NHID, 4,   PW2R);
  pack(W3l, NHID,  NHID, 4,   PW3L);  pack(W3r, NHID,  NHID, 4,   PW3R);
  pack(l1w, 256,   128,  8,   PL1);   pack(l2w, 128,   32,   4,   PL2);

  // ---- normalization scale, edge weights, readout accumulator ----
  k_colmax<<<12, 256, 0, stream>>>(x, NN1, FEATD, CM);
  EL(k_scl, KP1, CM, SCL, KP1);
  EL(k_fillf, EDG, EM0, 1.f, EDG);
  EL(k_fillf, NB * 256, HSUM, 0.f, NB * 256);

  // ================= layer 1 =================
  k_gemm<FEATD, 4, true, true><<<dim3(NN1 / 16, 2), 32, 0, stream>>>(
      x, SCL, PW1L, PW1R, H, R, NHID, 8);
  EL(k_fillf, NN1 * NHID, S, 0.f, NN1 * NHID);
  EL(k_fillf, NN1, CNT, 0.f, NN1);
  EL(k_edge_agg, EDG * 32, H, src0, dst0, EM0, S, CNT, EDG);
  EL(k_combine, NN1 * NHID, S, CNT, b1l, R, NN1 * NHID);
  // SAGPool 1
  EL(k_dot128, NN1, S, Wp1, HS, NN1);
  EL(k_fillf, NN1, DEGv, 1.f, NN1);
  EL(k_edge_deg, EDG, dst0, EM0, DEGv, EDG);
  EL(k_fillf, NN1, AGG, 0.f, NN1);
  EL(k_edge_sc, EDG, HS, src0, dst0, EM0, DEGv, AGG, EDG);
  EL(k_score_fin, NN1, AGG, HS, DEGv, bp1, SCR, NN1);
  EL(k_topk, NN1, SCR, N0, KK1, PERM, NN1);
  EL(k_filli, NN1, INV, -1, NN1);
  EL(k_scatter, NKK1, PERM, INV, NKK1);
  EL(k_pool, NKK1 * NHID, S, SCR, PERM, XP1, NKK1 * NHID);
  EL(k_remap, EDG, src0, dst0, EM0, INV, SRCA, DSTA, EMA, EDG);
  EL(k_readout, NB * NHID, XP1, KK1, HSUM, NB * NHID);

  // ================= layer 2 =================
  k_gemm<NHID, 4, true, false><<<dim3(NKK1 / 16, 2), 32, 0, stream>>>(
      XP1, nullptr, PW2L, PW2R, H, R, NHID, 8);
  EL(k_fillf, NKK1 * NHID, S, 0.f, NKK1 * NHID);
  EL(k_fillf, NKK1, CNT, 0.f, NKK1);
  EL(k_edge_agg, EDG * 32, H, SRCA, DSTA, EMA, S, CNT, EDG);
  EL(k_combine, NKK1 * NHID, S, CNT, b2l, R, NKK1 * NHID);
  EL(k_dot128, NKK1, S, Wp2, HS, NKK1);
  EL(k_fillf, NKK1, DEGv, 1.f, NKK1);
  EL(k_edge_deg, EDG, DSTA, EMA, DEGv, EDG);
  EL(k_fillf, NKK1, AGG, 0.f, NKK1);
  EL(k_edge_sc, EDG, HS, SRCA, DSTA, EMA, DEGv, AGG, EDG);
  EL(k_score_fin, NKK1, AGG, HS, DEGv, bp2, SCR, NKK1);
  EL(k_topk, NKK1, SCR, KK1, KK2, PERM, NKK1);
  EL(k_filli, NKK1, INV, -1, NKK1);
  EL(k_scatter, NKK2, PERM, INV, NKK2);
  EL(k_pool, NKK2 * NHID, S, SCR, PERM, XP2, NKK2 * NHID);
  EL(k_remap, EDG, SRCA, DSTA, EMA, INV, SRCB, DSTB, EMB, EDG);
  EL(k_readout, NB * NHID, XP2, KK2, HSUM, NB * NHID);

  // ================= layer 3 =================
  k_gemm<NHID, 4, true, false><<<dim3(NKK2 / 16, 2), 32, 0, stream>>>(
      XP2, nullptr, PW3L, PW3R, H, R, NHID, 8);
  EL(k_fillf, NKK2 * NHID, S, 0.f, NKK2 * NHID);
  EL(k_fillf, NKK2, CNT, 0.f, NKK2);
  EL(k_edge_agg, EDG * 32, H, SRCB, DSTB, EMB, S, CNT, EDG);
  EL(k_combine, NKK2 * NHID, S, CNT, b3l, R, NKK2 * NHID);
  EL(k_dot128, NKK2, S, Wp3, HS, NKK2);
  EL(k_fillf, NKK2, DEGv, 1.f, NKK2);
  EL(k_edge_deg, EDG, DSTB, EMB, DEGv, EDG);
  EL(k_fillf, NKK2, AGG, 0.f, NKK2);
  EL(k_edge_sc, EDG, HS, SRCB, DSTB, EMB, DEGv, AGG, EDG);
  EL(k_score_fin, NKK2, AGG, HS, DEGv, bp3, SCR, NKK2);
  EL(k_topk, NKK2, SCR, KK2, KK3, PERM, NKK2);
  EL(k_filli, NKK2, INV, -1, NKK2);
  EL(k_scatter, NKK3, PERM, INV, NKK3);
  EL(k_pool, NKK3 * NHID, S, SCR, PERM, XP3, NKK3 * NHID);
  EL(k_readout, NB * NHID, XP3, KK3, HSUM, NB * NHID);

  // ================= MLP head =================
  k_gemm<256, 4, false, false><<<dim3(NB / 16, 2), 32, 0, stream>>>(
      HSUM, nullptr, PL1, nullptr, H1, nullptr, NHID, 8);
  EL(k_bias_relu, NB * NHID, H1, l1b, NB * NHID, NHID);
  k_gemm<128, 2, false, false><<<dim3(NB / 16, 1), 32, 0, stream>>>(
      H1, nullptr, PL2, nullptr, out, nullptr, 32, 2);
  EL(k_bias_relu, NB * 32, out, l2b, NB * 32, 32);      // features -> d_out[0:1024]
  k_final<<<1, 32, 0, stream>>>(out, l3w, l3b, out + NB * 32);  // d_out[1024:1056]
}
